// NonAutoregressiveDecoder_54829552501260
// MI455X (gfx1250) — compile-verified, matrix operand order checked by reading, not run on hardware
//
#include <hip/hip_runtime.h>

typedef __attribute__((ext_vector_type(16))) _Float16 v16h;
typedef __attribute__((ext_vector_type(8)))  float    v8f;

#define KN 1024   // nodes
#define HD 128    // hidden

// Fast GELU (tanh form), algebraically reduced:
//   gelu(x) = 0.5x(1+tanh(u)) = x - x * rcp(exp(2u)+1),  2u = x*(c1 + c2*x^2)
// 5 VALU + 2 TRANS per element. Saturates correctly at +/-inf.
__device__ __forceinline__ float fast_gelu(float x) {
    float u2 = x * fmaf(x * x, 0.0713548162726f, 1.5957691216057308f);
    float e  = __expf(u2);
    float r  = __builtin_amdgcn_rcpf(e + 1.0f);
    return fmaf(-x, r, x);
}

// ---------------------------------------------------------------------------
// Prep 1: pi[i,h] = z[i]@eW1[:H][:,h] ; pjb[i,h] = z[i]@eW1[H:][:,h] + eb1[h]
// ---------------------------------------------------------------------------
__global__ void prep_pi_kernel(const float* __restrict__ z,
                               const float* __restrict__ eW1,
                               const float* __restrict__ eb1,
                               float* __restrict__ pi,
                               float* __restrict__ pjb) {
    int i = blockIdx.x;
    int h = threadIdx.x;
    const float* zr = z + i * HD;
    float s1 = 0.f, s2 = 0.f;
#pragma unroll 4
    for (int k = 0; k < HD; ++k) {
        float zv = zr[k];
        s1 = fmaf(zv, eW1[k * HD + h], s1);
        s2 = fmaf(zv, eW1[(HD + k) * HD + h], s2);
    }
    pi[i * HD + h]  = s1;
    pjb[i * HD + h] = s2 + eb1[h];
}

// ---------------------------------------------------------------------------
// Prep 2: coords_pred = gelu(z@cW1+cb1)@cW2 + cb2  -> d_out[0:2048]
// ---------------------------------------------------------------------------
__global__ void prep_coords_kernel(const float* __restrict__ z,
                                   const float* __restrict__ cW1,
                                   const float* __restrict__ cb1,
                                   const float* __restrict__ cW2,
                                   const float* __restrict__ cb2,
                                   float* __restrict__ out) {
    int i = blockIdx.x * blockDim.x + threadIdx.x;
    if (i >= KN) return;
    const float* zr = z + i * HD;
    float o0 = cb2[0], o1 = cb2[1];
    for (int c = 0; c < 64; ++c) {
        float s = cb1[c];
#pragma unroll 4
        for (int k = 0; k < HD; ++k) s = fmaf(zr[k], cW1[k * 64 + c], s);
        float g = fast_gelu(s);
        o0 = fmaf(g, cW2[c * 2 + 0], o0);
        o1 = fmaf(g, cW2[c * 2 + 1], o1);
    }
    out[i * 2 + 0] = o0;
    out[i * 2 + 1] = o1;
}

// ---------------------------------------------------------------------------
// Prep 3: swizzle eW2 (128x64 f32) into per-fragment f16 B layout.
// B 16-bit 32x16 layout (mirror of A): n = nt*16 + (lane&15),
//   k = kk*32 + t + 8*(t>=8) + 8*(lane>=16)
// ---------------------------------------------------------------------------
__global__ void prep_w2frag_kernel(const float* __restrict__ eW2,
                                   _Float16* __restrict__ w2f) {
    int id = blockIdx.x * blockDim.x + threadIdx.x;   // 0..8191
    int t  = id & 15;
    int l  = (id >> 4) & 31;
    int f  = id >> 9;
    int kk = f >> 2, nt = f & 3;
    int k  = kk * 32 + t + (t >= 8 ? 8 : 0) + ((l >= 16) ? 8 : 0);
    int n  = nt * 16 + (l & 15);
    w2f[id] = (_Float16)eW2[k * 64 + n];
}

// ---------------------------------------------------------------------------
// Main fused kernel: per wave, 16 pairs (i0..i0+15, fixed j) at a time.
//   A (16x32 f16) = gelu(pi[i,h] + pjb[j,h]) generated in registers
//   acc[nt] (16x16 f32) += A x B[kk][nt] via v_wmma_f32_16x16x32_f16
//   adj_raw[i,j] = sum_c gelu(acc + eb2[c]) * eW3[c] + eb3  (butterfly reduce)
// pj DS loads are software-pipelined one k-step ahead.
// ---------------------------------------------------------------------------
__global__ void __launch_bounds__(256) edge_main_kernel(
        const float* __restrict__ pi, const float* __restrict__ pjb,
        const _Float16* __restrict__ w2f, const float* __restrict__ eb2,
        const float* __restrict__ eW3, const float* __restrict__ eb3,
        float* __restrict__ adj) {
    __shared__ __align__(16) float    s_pi[16 * 132];   // padded stride: conflict-free
    __shared__ __align__(32) _Float16 s_w2[16 * 512];   // 16 frags x 32 lanes x 16 halfs
    __shared__ __align__(16) float    s_pj[64 * HD];    // this block's 64 pjb rows

    int tid = threadIdx.x;
    int i0  = blockIdx.x * 16;
    int j0  = blockIdx.y * 64;

    for (int e = tid; e < 16 * HD; e += 256) {
        int r = e >> 7, c = e & 127;
        s_pi[r * 132 + c] = pi[(i0 + r) * HD + c];
    }
    {
        const uint32_t* src = reinterpret_cast<const uint32_t*>(w2f);
        uint32_t*       dst = reinterpret_cast<uint32_t*>(s_w2);
        for (int e = tid; e < 4096; e += 256) dst[e] = src[e];
    }
    {
        const float4* src = reinterpret_cast<const float4*>(pjb + j0 * HD);
        float4*       dst = reinterpret_cast<float4*>(s_pj);
        for (int e = tid; e < (64 * HD) / 4; e += 256) dst[e] = src[e];
    }
    __syncthreads();

    int lane = tid & 31;
    int wave = tid >> 5;
    int lrow = lane & 15;        // A row M / B,C,D column N
    int lhi  = lane >> 4;        // K-half select (A/B), +8 row offset (C/D)
    const float* pirow = &s_pi[lrow * 132];

    float b2v[4], w3v[4];
#pragma unroll
    for (int nt = 0; nt < 4; ++nt) {
        b2v[nt] = eb2[nt * 16 + lrow];
        w3v[nt] = eW3[nt * 16 + lrow];
    }
    float b3 = eb3[0];

    for (int jj = 0; jj < 8; ++jj) {
        int jl = wave * 8 + jj;
        int j  = j0 + jl;
        const float* pjrow = &s_pj[jl * HD + lhi * 8];

        v8f acc[4] = {};

        // Prologue: load k-step 0 pj fragments.
        float4 q0, q1, q2, q3;
        {
            const float4* pj4 = reinterpret_cast<const float4*>(pjrow);
            q0 = pj4[0]; q1 = pj4[1]; q2 = pj4[4]; q3 = pj4[5];
        }

#pragma unroll
        for (int kk = 0; kk < 4; ++kk) {
            // Issue next k-step's pj loads before this step's compute.
            float4 n0, n1, n2, n3;
            if (kk < 3) {
                const float4* pj4n =
                    reinterpret_cast<const float4*>(pjrow + (kk + 1) * 32);
                n0 = pj4n[0]; n1 = pj4n[1]; n2 = pj4n[4]; n3 = pj4n[5];
            }

            int hb = kk * 32 + lhi * 8;
            const float4* pi4 = reinterpret_cast<const float4*>(pirow + hb);
            float4 p0 = pi4[0], p1 = pi4[1], p2 = pi4[4], p3 = pi4[5];

            float piv[16] = {p0.x,p0.y,p0.z,p0.w, p1.x,p1.y,p1.z,p1.w,
                             p2.x,p2.y,p2.z,p2.w, p3.x,p3.y,p3.z,p3.w};
            float pjv[16] = {q0.x,q0.y,q0.z,q0.w, q1.x,q1.y,q1.z,q1.w,
                             q2.x,q2.y,q2.z,q2.w, q3.x,q3.y,q3.z,q3.w};

            v16h a;
#pragma unroll
            for (int t = 0; t < 16; ++t)
                a[t] = (_Float16)fast_gelu(piv[t] + pjv[t]);

#pragma unroll
            for (int nt = 0; nt < 4; ++nt) {
                const v16h b = *reinterpret_cast<const v16h*>(
                    &s_w2[((kk * 4 + nt) * 32 + lane) * 16]);
                acc[nt] = __builtin_amdgcn_wmma_f32_16x16x32_f16(
                    false, a, false, b, (short)0, acc[nt], false, false);
            }

            if (kk < 3) { q0 = n0; q1 = n1; q2 = n2; q3 = n3; }
        }

        // Epilogue: gelu(h2)*eW3, butterfly-reduce 16 columns per row.
#pragma unroll
        for (int r = 0; r < 8; ++r) {
            float v = 0.f;
#pragma unroll
            for (int nt = 0; nt < 4; ++nt)
                v = fmaf(fast_gelu(acc[nt][r] + b2v[nt]), w3v[nt], v);
            v += __shfl_xor(v, 1, 32);
            v += __shfl_xor(v, 2, 32);
            v += __shfl_xor(v, 4, 32);
            v += __shfl_xor(v, 8, 32);
            if (lrow == 0)
                adj[(i0 + r + lhi * 8) * KN + j] = v + b3;
        }
    }
}

// ---------------------------------------------------------------------------
// In-place symmetrization: adj = (adj + adj^T)/2. Each i<j pair touched once.
// ---------------------------------------------------------------------------
__global__ void symmetrize_kernel(float* __restrict__ adj) {
    int i = blockIdx.x;
    int j = blockIdx.y * blockDim.x + threadIdx.x;
    if (j <= i || j >= KN) return;
    float a = adj[i * KN + j];
    float b = adj[j * KN + i];
    float m = 0.5f * (a + b);
    adj[i * KN + j] = m;
    adj[j * KN + i] = m;
}

extern "C" void kernel_launch(void* const* d_in, const int* in_sizes, int n_in,
                              void* d_out, int out_size, void* d_ws, size_t ws_size,
                              hipStream_t stream) {
    const float* z   = (const float*)d_in[0];
    const float* cW1 = (const float*)d_in[1];
    const float* cb1 = (const float*)d_in[2];
    const float* cW2 = (const float*)d_in[3];
    const float* cb2 = (const float*)d_in[4];
    const float* eW1 = (const float*)d_in[5];
    const float* eb1 = (const float*)d_in[6];
    const float* eW2 = (const float*)d_in[7];
    const float* eb2 = (const float*)d_in[8];
    const float* eW3 = (const float*)d_in[9];
    const float* eb3 = (const float*)d_in[10];

    float* out    = (float*)d_out;
    float* coords = out;          // [1024,2]
    float* adj    = out + 2048;   // [1024,1024]

    // Workspace: pi (512KB) | pjb (512KB) | w2frag f16 (16KB)
    float*    pi  = (float*)d_ws;
    float*    pjb = pi + KN * HD;
    _Float16* w2f = (_Float16*)(pjb + KN * HD);

    prep_pi_kernel    <<<dim3(KN),     dim3(HD),  0, stream>>>(z, eW1, eb1, pi, pjb);
    prep_coords_kernel<<<dim3(4),      dim3(256), 0, stream>>>(z, cW1, cb1, cW2, cb2, coords);
    prep_w2frag_kernel<<<dim3(32),     dim3(256), 0, stream>>>(eW2, w2f);
    edge_main_kernel  <<<dim3(64, 16), dim3(256), 0, stream>>>(pi, pjb, w2f, eb2, eW3, eb3, adj);
    symmetrize_kernel <<<dim3(KN, 4),  dim3(256), 0, stream>>>(adj);
}
